// NAS_SLSTM_31808527794713
// MI455X (gfx1250) — matrix operand order repaired and checked by reading, not compiled, not run on hardware
//
#include <hip/hip_runtime.h>
#include <hip/hip_bf16.h>
#include <cstddef>

// Shapes from the reference
#define B_ 64
#define T_ 512
#define I_ 256
#define H_ 512
#define G_ 2048   // 4*H

typedef __attribute__((ext_vector_type(16))) __bf16 v16bf;
typedef __attribute__((ext_vector_type(8)))  float  v8f;

#if __has_builtin(__builtin_amdgcn_tensor_load_to_lds)
#define HAVE_TDM 1
typedef __attribute__((ext_vector_type(4))) unsigned int u32x4;
typedef __attribute__((ext_vector_type(8))) int i32x8;
typedef __attribute__((ext_vector_type(4))) int i32x4;
#else
#define HAVE_TDM 0
#endif

// Fragment = 16 bf16 values per lane, loaded as two 16-byte LDS reads.
union Frag {
    uint4 q[2];
    v16bf v;
};

__device__ __forceinline__ unsigned short f2bf(float f) {
    unsigned int u = __builtin_bit_cast(unsigned int, f);
    u += 0x7FFFu + ((u >> 16) & 1u);          // round-to-nearest-even
    return (unsigned short)(u >> 16);
}

__device__ __forceinline__ float sigmoidf_(float x) {
    return 1.0f / (1.0f + __expf(-x));
}

// ---------------------------------------------------------------------------
// Phase 1: x_proj[t*B+b][g] = sum_i x[b][t][i] * W_ih[i][g] + bias[g]
// GEMM M=32768, K=256, N=2048. WG tile 64(M) x 128(N); 8 waves in 2x4 grid,
// each wave owns a 32x32 block = 4 WMMA accumulators.
// ---------------------------------------------------------------------------
__global__ __launch_bounds__(256) void xproj_gemm_kernel(
    const float* __restrict__ x, const float* __restrict__ w_ih,
    const float* __restrict__ bias, float* __restrict__ xp)
{
    __shared__ unsigned short ldsA[64 * 32];        // [m][k] bf16, row-major
    __shared__ unsigned short ldsBf[8 * 32 * 16];   // [colgrp][lane][e] bf16

    const int tid  = threadIdx.x;
    const int lane = tid & 31;
    const int wid  = tid >> 5;
    const int lm   = lane & 15;
    const bool hiL = lane >= 16;
    const int waveM = wid & 1;        // 0..1 -> 32-row half
    const int waveN = wid >> 1;       // 0..3 -> 32-col group
    const int Mbase = blockIdx.y * 64;
    const int Nbase = blockIdx.x * 128;

    v8f acc[2][2] = {};
    for (int k0 = 0; k0 < I_; k0 += 32) {
        // Stage A: 64x32 (x rows are (b,t); output row m = t*B + b)
        #pragma unroll
        for (int i = 0; i < 8; ++i) {
            int e = tid + i * 256;
            int r = e >> 5, c = e & 31;
            int m = Mbase + r;
            int b = m & (B_ - 1);
            int tq = m >> 6;
            ldsA[e] = f2bf(x[((size_t)b * T_ + tq) * I_ + k0 + c]);
        }
        // Stage B: 32x128 of W_ih, coalesced global read, fragment-order store.
        #pragma unroll
        for (int i = 0; i < 16; ++i) {
            int e = tid + i * 256;
            int r = e >> 7, c = e & 127;
            int ln = (r < 16) ? (c & 15) : (16 + (c & 15));
            int ee = r & 15;
            int w  = c >> 4;
            ldsBf[((w * 32 + ln) << 4) + ee] =
                f2bf(w_ih[(size_t)(k0 + r) * G_ + Nbase + c]);
        }
        __syncthreads();

        #pragma unroll
        for (int mi = 0; mi < 2; ++mi) {
            Frag af;
            const unsigned short* ap =
                &ldsA[(waveM * 32 + mi * 16 + lm) * 32 + (hiL ? 16 : 0)];
            af.q[0] = *(const uint4*)ap;
            af.q[1] = *(const uint4*)(ap + 8);
            #pragma unroll
            for (int ni = 0; ni < 2; ++ni) {
                Frag bf;
                const unsigned short* bp = &ldsBf[((waveN * 2 + ni) * 32 + lane) << 4];
                bf.q[0] = *(const uint4*)bp;
                bf.q[1] = *(const uint4*)(bp + 8);
                acc[mi][ni] = __builtin_amdgcn_wmma_f32_16x16x32_bf16(
                    false, af.v, false, bf.v, (short)0, acc[mi][ni], false, false);
            }
        }
        __syncthreads();
    }

    #pragma unroll
    for (int ni = 0; ni < 2; ++ni) {
        const int g  = Nbase + (waveN * 2 + ni) * 16 + lm;
        const float bv = bias[g];
        #pragma unroll
        for (int mi = 0; mi < 2; ++mi) {
            #pragma unroll
            for (int r = 0; r < 8; ++r) {
                int m = Mbase + waveM * 32 + mi * 16 + r + (hiL ? 8 : 0);
                xp[(size_t)m * G_ + g] = acc[mi][ni][r] + bv;
            }
        }
    }
}

// ---------------------------------------------------------------------------
// Phase 2: persistent recurrence. 64 WGs = 4 B-tiles x 16 H-slices.
// W_hh slice (512x128) lives in LDS as bf16 in WMMA fragment order (one-time
// swizzle); per-step h broadcast arrives via the Tensor Data Mover.
// ---------------------------------------------------------------------------
__global__ __launch_bounds__(256) void slstm_recurrence_kernel(
    const float* __restrict__ w_hh, const float* __restrict__ xp,
    unsigned short* __restrict__ h_ws, unsigned int* __restrict__ cnt,
    float* __restrict__ out)
{
    extern __shared__ char smem[];
    unsigned short* ldsWf = (unsigned short*)smem;                 // [16][8][32][16] bf16
    unsigned short* ldsH  = (unsigned short*)(smem + 131072);      // [16][512] bf16
    float*          ldsG  = (float*)(smem + 131072 + 16384);       // [16][128] f32

    const int tid  = threadIdx.x;
    const int lane = tid & 31;
    const int wid  = tid >> 5;
    const int lm   = lane & 15;
    const bool hiL = lane >= 16;
    const int b0    = (blockIdx.x & 3) * 16;
    const int hbase = (blockIdx.x >> 2) * 32;
    const int wcol  = wid * 16 + lm;

    // One-time: W_hh slice -> LDS in fragment order (coalesced global reads).
    for (int i = 0; i < 256; ++i) {
        int idx = tid + i * 256;                 // 65536 elements
        int k = idx >> 7, c = idx & 127;         // k: 0..511, c: 0..127
        int kt = k >> 5, r = k & 31;
        int ln = (r < 16) ? (c & 15) : (16 + (c & 15));
        int ee = r & 15;
        int w  = c >> 4;
        int g  = c >> 5, j = c & 31;
        ldsWf[(((kt * 8 + w) * 32 + ln) << 4) + ee] =
            f2bf(w_hh[(size_t)k * G_ + g * H_ + hbase + j]);
    }

    // h0 = 0 for our slice (packed 32-bit stores); c kept in registers.
    {
        int r = tid >> 4, j2 = (tid & 15) * 2;
        *(unsigned int*)&h_ws[(size_t)(b0 + r) * H_ + hbase + j2] = 0u;
    }
    float creg0 = 0.0f, creg1 = 0.0f;

#if HAVE_TDM
    // Tensor DMA descriptor: 1-D copy of 16 KB (2048 x 8B) h rows -> ldsH.
    u32x4 g0; i32x8 g1; i32x4 g2, g3; i32x8 g4;
    {
        unsigned long long ga = (unsigned long long)(h_ws + (size_t)b0 * H_);
        unsigned ldsAddr = (unsigned)__builtin_amdgcn_groupstaticsize() + 131072u;
        g0[0] = 1u;                                              // count = 1 (valid D#)
        g0[1] = ldsAddr;                                         // lds_addr
        g0[2] = (unsigned)(ga & 0xFFFFFFFFu);                    // global_addr lo
        g0[3] = (unsigned)((ga >> 32) & 0x1FFFFFFu) | 0x80000000u; // addr hi | type=2
        g1[0] = 3 << 16;                  // workgroup_mask=0, data_size=3 (8 bytes)
        g1[1] = 0x800 << 16;              // tensor_dim0 = 2048 (lo16 at bits 63:48)
        g1[2] = 0;                        // tensor_dim0 hi = 0, tensor_dim1 = 0
        g1[3] = 0x800 << 16;              // tile_dim0 = 2048 (bits 127:112)
        g1[4] = 0;                        // tile_dim1/2 unused (1-D)
        g1[5] = 0x800;                    // tensor_dim0_stride = 2048
        g1[6] = 0;
        g1[7] = 0;
        g2 = (i32x4){0, 0, 0, 0};
        g3 = (i32x4){0, 0, 0, 0};
        g4 = (i32x8){0, 0, 0, 0, 0, 0, 0, 0};
    }
#endif

    unsigned int gen = 0;
    auto gbar = [&]() {
        __syncthreads();
        __threadfence();
        ++gen;
        if (tid == 0) {
            atomicAdd(cnt, 1u);
            const unsigned int target = 64u * gen;
            while (__hip_atomic_load(cnt, __ATOMIC_ACQUIRE,
                                     __HIP_MEMORY_SCOPE_AGENT) < target) {
                __builtin_amdgcn_s_sleep(1);
            }
        }
        __syncthreads();
    };
    gbar();   // all h0 slices visible

    const int er = tid >> 4;            // elementwise row 0..15
    const int ej = (tid & 15) * 2;      // elementwise col pair (even)

    for (int t = 0; t < T_; ++t) {
        // Stage h rows for our B-tile (16 KB contiguous).
#if HAVE_TDM
        if (tid < 32) {                 // one wave issues the DMA; EXEC ignored
            __builtin_amdgcn_tensor_load_to_lds(g0, g1, g2, g3, g4, 0);
#if __has_builtin(__builtin_amdgcn_s_wait_tensorcnt)
            __builtin_amdgcn_s_wait_tensorcnt(0);
#else
            asm volatile("s_wait_tensorcnt 0x0" ::: "memory");
#endif
        }
        __syncthreads();
#else
        {
            uint4* dst = (uint4*)ldsH;
            const uint4* src = (const uint4*)(h_ws + (size_t)b0 * H_);
            #pragma unroll
            for (int i = 0; i < 4; ++i) dst[tid + i * 256] = src[tid + i * 256];
        }
        __syncthreads();
#endif

        v8f acc = {};
        #pragma unroll 4
        for (int kt = 0; kt < 16; ++kt) {
            Frag af, bf;
            const unsigned short* ap = &ldsH[lm * 512 + kt * 32 + (hiL ? 16 : 0)];
            af.q[0] = *(const uint4*)ap;
            af.q[1] = *(const uint4*)(ap + 8);
            const unsigned short* bp = &ldsWf[((kt * 8 + wid) * 32 + lane) << 4];
            bf.q[0] = *(const uint4*)bp;
            bf.q[1] = *(const uint4*)(bp + 8);
            acc = __builtin_amdgcn_wmma_f32_16x16x32_bf16(
                false, af.v, false, bf.v, (short)0, acc, false, false);
        }

        // Park raw gate pre-activations so every thread can gather all 4 gates.
        #pragma unroll
        for (int r = 0; r < 8; ++r) {
            int M = r + (hiL ? 8 : 0);
            ldsG[M * 128 + wcol] = acc[r];
        }
        __syncthreads();

        // Gate nonlinearities + state update, float2-wide (2 elements/thread).
        {
            size_t xb = ((size_t)t * B_ + b0 + er) * (size_t)G_ + hbase + ej;
            float2 xi = *(const float2*)&xp[xb];
            float2 xf = *(const float2*)&xp[xb + 512];
            float2 xc = *(const float2*)&xp[xb + 1024];
            float2 xo = *(const float2*)&xp[xb + 1536];
            if (t + 1 < T_) {   // prefetch next step's x_proj rows
                __builtin_prefetch(&xp[xb + (size_t)B_ * G_], 0, 1);
                __builtin_prefetch(&xp[xb + (size_t)B_ * G_ + 1024], 0, 1);
            }
            float2 gi = *(const float2*)&ldsG[er * 128 +      ej];
            float2 gf = *(const float2*)&ldsG[er * 128 + 32 + ej];
            float2 gc = *(const float2*)&ldsG[er * 128 + 64 + ej];
            float2 go = *(const float2*)&ldsG[er * 128 + 96 + ej];

            float ig0 = sigmoidf_(gi.x + xi.x), ig1 = sigmoidf_(gi.y + xi.y);
            float fg0 = sigmoidf_(gf.x + xf.x), fg1 = sigmoidf_(gf.y + xf.y);
            float cg0 = tanhf(gc.x + xc.x),     cg1 = tanhf(gc.y + xc.y);
            float og0 = go.x + xo.x,            og1 = go.y + xo.y;

            creg0 = fg0 * creg0 + ig0 * cg0;
            creg1 = fg1 * creg1 + ig1 * cg1;
            float hy0 = og0 * tanhf(creg0) + og0;   // reference: no sigmoid on og
            float hy1 = og1 * tanhf(creg1) + og1;

            float2 hy2; hy2.x = hy0; hy2.y = hy1;
            *(float2*)&out[((size_t)(b0 + er) * T_ + t) * H_ + hbase + ej] = hy2;
            unsigned int packed =
                ((unsigned int)f2bf(hy1) << 16) | (unsigned int)f2bf(hy0);
            *(unsigned int*)&h_ws[(size_t)(b0 + er) * H_ + hbase + ej] = packed;
        }
        gbar();   // publish h(t) to all WGs before step t+1
    }
}

// ---------------------------------------------------------------------------
extern "C" void kernel_launch(void* const* d_in, const int* in_sizes, int n_in,
                              void* d_out, int out_size, void* d_ws, size_t ws_size,
                              hipStream_t stream) {
    const float* x    = (const float*)d_in[0];
    const float* w_ih = (const float*)d_in[1];
    const float* w_hh = (const float*)d_in[2];
    const float* bias = (const float*)d_in[3];
    float* out = (float*)d_out;

    const size_t XPROJ_BYTES = (size_t)T_ * B_ * G_ * sizeof(float);   // 256 MB
    const size_t H_BYTES     = (size_t)B_ * H_ * sizeof(unsigned short);
    float*          xp   = (float*)d_ws;
    unsigned short* h_ws = (unsigned short*)((char*)d_ws + XPROJ_BYTES);
    unsigned int*   cnt  = (unsigned int*)((char*)d_ws + XPROJ_BYTES + H_BYTES);

    // Phase 1: input projection GEMM (+bias) into workspace
    dim3 g1(G_ / 128, (B_ * T_) / 64);
    xproj_gemm_kernel<<<g1, 256, 0, stream>>>(x, w_ih, bias, xp);

    // Reset the grid-barrier counter each call (graph-capture safe)
    (void)hipMemsetAsync(cnt, 0, 128, stream);

    // Phase 2: persistent recurrence, 64 WGs, 152 KB dynamic LDS each
    const size_t smem2 = 131072 + 16384 + (size_t)16 * 128 * sizeof(float);
    slstm_recurrence_kernel<<<dim3(64), 256, smem2, stream>>>(w_hh, xp, h_ws, cnt, out);
}
